// IA_76459007803640
// MI455X (gfx1250) — matrix operand order repaired and checked by reading, not compiled, not run on hardware
//
#include <hip/hip_runtime.h>
#include <hip/hip_bf16.h>

typedef __attribute__((ext_vector_type(16))) _Float16 v16h;
typedef __attribute__((ext_vector_type(8)))  float    v8f;

#define LL 64
#define DD 256
#define EPSF 1e-5f

// ---------------------------------------------------------------- helpers
__device__ __forceinline__ float mishf(float x) {
    float sp = (x > 20.f) ? x : log1pf(expf(x));
    return x * tanhf(sp);
}

__device__ __forceinline__ float block_sum256(float v, float* red) {
    int t = threadIdx.x;
    red[t] = v; __syncthreads();
    #pragma unroll
    for (int s = 128; s > 0; s >>= 1) {
        if (t < s) red[t] += red[t + s];
        __syncthreads();
    }
    float r = red[0]; __syncthreads();
    return r;
}

// A fragment (16x32 f16, row-major source, stride ld):
// lane = M (0..15); lanes>=16 use K offset +16; elements 0..15 = K 0..15(+kb)
__device__ __forceinline__ v16h frag_a_rm(const _Float16* p, int ld, int m0, int k0) {
    int lane = threadIdx.x & 31;
    int m  = lane & 15;
    int kb = (lane >> 4) << 4;             // 0 or 16
    const _Float16* r = p + (m0 + m) * ld + k0 + kb;
    v16h a;
    #pragma unroll
    for (int j = 0; j < 16; ++j) a[j] = r[j];
    return a;
}

// B fragment (32x16 f16) from K-major (row-major B[k][n]) source, stride ld
__device__ __forceinline__ v16h frag_b_km(const _Float16* p, int ld, int k0, int n0) {
    int lane = threadIdx.x & 31;
    int n  = lane & 15;
    int kb = (lane >> 4) << 4;             // 0 or 16
    const _Float16* c = p + (k0 + kb) * ld + n0 + n;
    v16h b;
    #pragma unroll
    for (int j = 0; j < 16; ++j) b[j] = c[j * ld];
    return b;
}

// B fragment (32x16 f16) where B = Xᵀ and X is row-major [n][k] with stride ld
// (contiguous 32-byte read per lane — ideal for S·Qᵀ and enh·Wᵀ)
__device__ __forceinline__ v16h frag_b_tr(const _Float16* p, int ld, int k0, int n0) {
    int lane = threadIdx.x & 31;
    int n  = lane & 15;
    int kb = (lane >> 4) << 4;             // 0 or 16
    const _Float16* r = p + (n0 + n) * ld + k0 + kb;
    v16h b;
    #pragma unroll
    for (int j = 0; j < 16; ++j) b[j] = r[j];
    return b;
}

// C/D 16x16 f32 store: lane = N, VGPR j = row M=j (+8 for lanes>=16)
__device__ __forceinline__ void store_c(float* p, int ld, int m0, int n0, v8f c) {
    int lane = threadIdx.x & 31;
    int n  = lane & 15;
    int mb = (lane >> 4) << 3;
    #pragma unroll
    for (int j = 0; j < 8; ++j) p[(m0 + mb + j) * ld + n0 + n] = c[j];
}

#define WMMA32(a, b, c) \
    __builtin_amdgcn_wmma_f32_16x16x32_f16(false, (a), false, (b), (short)0, (c), false, false)

// ---------------------------------------------------------------- kernel 1: LayerNorm rows of 256
__global__ __launch_bounds__(256) void ln_kernel(const float* __restrict__ x,
                                                 const float* __restrict__ g,
                                                 const float* __restrict__ b,
                                                 float* __restrict__ y32,
                                                 _Float16* __restrict__ y16) {
    __shared__ float red[256];
    size_t row = blockIdx.x;
    int d = threadIdx.x;
    float v = x[row * DD + d];
    red[d] = v; __syncthreads();
    #pragma unroll
    for (int s = 128; s > 0; s >>= 1) { if (d < s) red[d] += red[d + s]; __syncthreads(); }
    float mu = red[0] * (1.f / DD); __syncthreads();
    float c = v - mu;
    red[d] = c * c; __syncthreads();
    #pragma unroll
    for (int s = 128; s > 0; s >>= 1) { if (d < s) red[d] += red[d + s]; __syncthreads(); }
    float var = red[0] * (1.f / DD);
    float o = c * rsqrtf(var + EPSF) * g[d] + b[d];
    y32[row * DD + d] = o;
    y16[row * DD + d] = (_Float16)o;
}

// ---------------------------------------------------------------- kernel 2: f32 -> f16 convert
__global__ void cvt_kernel(const float* __restrict__ in, _Float16* __restrict__ out, int n) {
    int i = blockIdx.x * blockDim.x + threadIdx.x;
    if (i < n) out[i] = (_Float16)in[i];
}

// ---------------------------------------------------------------- kernel 3: one workgroup per pair
__global__ __launch_bounds__(256) void pair_kernel(
    const float* __restrict__ sn, const float* __restrict__ qn,
    const _Float16* __restrict__ shg, const _Float16* __restrict__ qhg,
    const _Float16* __restrict__ Wh, const float* __restrict__ proj_b,
    const float* __restrict__ ln2_g, const float* __restrict__ ln2_b,
    const float* __restrict__ W1, const float* __restrict__ b1,
    const float* __restrict__ W2, const float* __restrict__ b2,
    const float* __restrict__ W3, const float* __restrict__ b3,
    float* __restrict__ logits, int NQ, int NK)
{
    extern __shared__ char smem_raw[];
    _Float16* s_h  = (_Float16*)smem_raw;            // 64*256 f16   = 32768 B
    _Float16* q_h  = s_h + LL * DD;                  // 32768 B
    float*    att  = (float*)(q_h + LL * DD);        // 64*64 f32    = 16384 B
    _Float16* p1   = (_Float16*)(att + LL * LL);     // 8192 B  softmax over k (rows)
    _Float16* p2t  = p1 + LL * LL;                   // 8192 B  softmax over l, stored transposed
    _Float16* enh  = p2t + LL * LL;                  // 64*1024 f16  = 131072 B
    float*    pout = (float*)(enh + LL * 1024);      // 64*256 f32   = 65536 B
    float*    poolq = pout + LL * DD;                // 512 f32
    float*    pools = poolq + 512;                   // 512 f32
    float*    red   = pools + 512;                   // 256 f32

    const int m    = blockIdx.x;
    const int bidx = m / (NQ * NK);
    const int rem  = m % (NQ * NK);
    const int iq   = rem / NK;
    const int jj   = rem % NK;
    const size_t srow = (size_t)bidx * NK + jj;
    const size_t qrow = (size_t)bidx * NQ + iq;

    const int tid  = threadIdx.x;
    const int wave = tid >> 5;

    // ---- stage LN'd tiles (f16) into LDS via CDNA5 async memory->LDS DMA
    // (GLOBAL_LOAD_ASYNC_TO_LDS_B128: no VGPR round-trip, tracked with ASYNCcnt)
    {
        const uint4* g1 = (const uint4*)(shg + srow * LL * DD);
        const uint4* g2 = (const uint4*)(qhg + qrow * LL * DD);
        // low 32 bits of a generic pointer to LDS == LDS byte offset
        unsigned lb1 = (unsigned)(uintptr_t)s_h;
        unsigned lb2 = (unsigned)(uintptr_t)q_h;
        for (int i = tid; i < LL * DD / 8; i += 256) {
            unsigned lo1 = lb1 + (unsigned)i * 16u;
            unsigned lo2 = lb2 + (unsigned)i * 16u;
            asm volatile("global_load_async_to_lds_b128 %0, %1, off"
                         :: "v"(lo1), "v"(g1 + i) : "memory");
            asm volatile("global_load_async_to_lds_b128 %0, %1, off"
                         :: "v"(lo2), "v"(g2 + i) : "memory");
        }
        asm volatile("s_wait_asynccnt 0" ::: "memory");
    }
    __syncthreads();

    // ---- att = S @ Qᵀ   (64x64, K=256) : 16 tiles over 8 waves
    for (int t = wave; t < 16; t += 8) {
        int tm = (t >> 2) * 16, tn = (t & 3) * 16;
        v8f acc = {};
        for (int k0 = 0; k0 < DD; k0 += 32) {
            v16h a  = frag_a_rm(s_h, DD, tm, k0);
            v16h bb = frag_b_tr(q_h, DD, k0, tn);
            acc = WMMA32(a, bb, acc);
        }
        store_c(att, LL, tm, tn, acc);
    }
    __syncthreads();

    // ---- softmaxes (f32): rows (axis k) -> p1 ; cols (axis l) -> p2t (transposed)
    if (tid < LL) {
        int rI = tid;
        float mx = -1e30f;
        for (int k = 0; k < LL; ++k) mx = fmaxf(mx, att[rI * LL + k]);
        float sum = 0.f;
        for (int k = 0; k < LL; ++k) sum += expf(att[rI * LL + k] - mx);
        float inv = 1.f / sum;
        for (int k = 0; k < LL; ++k)
            p1[rI * LL + k] = (_Float16)(expf(att[rI * LL + k] - mx) * inv);
    } else if (tid < 2 * LL) {
        int cI = tid - LL;
        float mx = -1e30f;
        for (int l = 0; l < LL; ++l) mx = fmaxf(mx, att[l * LL + cI]);
        float sum = 0.f;
        for (int l = 0; l < LL; ++l) sum += expf(att[l * LL + cI] - mx);
        float inv = 1.f / sum;
        for (int l = 0; l < LL; ++l)
            p2t[cI * LL + l] = (_Float16)(expf(att[l * LL + cI] - mx) * inv);
    }
    __syncthreads();

    // ---- two symmetric branches: br 0 = support side, br 1 = query side
    for (int br = 0; br < 2; ++br) {
        const _Float16* P  = (br == 0) ? p1 : p2t;
        const _Float16* Bm = (br == 0) ? q_h : s_h;                 // attended values
        const float*    xg = (br == 0) ? sn + srow * LL * DD
                                       : qn + qrow * LL * DD;       // f32 original
        float* pool = (br == 0) ? pools : poolq;

        // attended = P @ Bm  (64x256, K=64) -> pout
        for (int t = wave; t < 64; t += 8) {
            int tm = (t >> 4) * 16, tn = (t & 15) * 16;
            v8f acc = {};
            for (int k0 = 0; k0 < LL; k0 += 32) {
                v16h a  = frag_a_rm(P, LL, tm, k0);
                v16h bb = frag_b_km(Bm, DD, k0, tn);
                acc = WMMA32(a, bb, acc);
            }
            store_c(pout, DD, tm, tn, acc);
        }
        __syncthreads();

        // enh = [x, att, |x-att|, x*att] as f16 [64 x 1024]
        for (int i = tid; i < LL * DD; i += 256) {
            int l = i >> 8, d = i & 255;
            float x = xg[i];
            float a = pout[i];
            _Float16* e = enh + (size_t)l * 1024;
            e[d]       = (_Float16)x;
            e[256 + d] = (_Float16)a;
            e[512 + d] = (_Float16)fabsf(x - a);
            e[768 + d] = (_Float16)(x * a);
        }
        __syncthreads();

        // proj: pout = mish(enh @ Whᵀ + proj_b)   (64x256, K=1024)
        for (int t = wave; t < 64; t += 8) {
            int tm = (t >> 4) * 16, tn = (t & 15) * 16;
            v8f acc = {};
            for (int k0 = 0; k0 < 1024; k0 += 32) {
                v16h a  = frag_a_rm(enh, 1024, tm, k0);
                v16h bb = frag_b_tr(Wh, 1024, k0, tn);
                acc = WMMA32(a, bb, acc);
            }
            float bias = proj_b[tn + (tid & 15)];
            #pragma unroll
            for (int j = 0; j < 8; ++j) acc[j] = mishf(acc[j] + bias);
            store_c(pout, DD, tm, tn, acc);
        }
        __syncthreads();

        // pool (max+mean over L) then LayerNorm over 512
        {
            int d = tid;
            float mx = -1e30f, sm = 0.f;
            for (int l = 0; l < LL; ++l) {
                float v = pout[l * DD + d];
                mx = fmaxf(mx, v); sm += v;
            }
            float mean = sm * (1.f / LL);
            float ssum = block_sum256(mx + mean, red);
            float ssq  = block_sum256(mx * mx + mean * mean, red);
            float mu  = ssum * (1.f / 512.f);
            float var = ssq * (1.f / 512.f) - mu * mu;
            float inv = rsqrtf(var + EPSF);
            pool[d]      = (mx   - mu) * inv * ln2_g[d]      + ln2_b[d];
            pool[DD + d] = (mean - mu) * inv * ln2_g[DD + d] + ln2_b[DD + d];
        }
        __syncthreads();
    }

    // ---- head: cat = [q_pool, s_pool]; logit = mish(mish(cat@W1ᵀ+b1)@W2ᵀ+b2)*W3+b3
    {
        int d = tid;
        const float* w1r = W1 + (size_t)d * 1024;
        float acc = b1[d];
        for (int k = 0; k < 512; ++k) acc = fmaf(w1r[k],       poolq[k], acc);
        for (int k = 0; k < 512; ++k) acc = fmaf(w1r[512 + k], pools[k], acc);
        float part = mishf(acc) * W2[d];
        float hs = block_sum256(part, red);
        if (tid == 0)
            logits[m] = mishf(hs + b2[0]) * W3[0] + b3[0];
    }
}

// ---------------------------------------------------------------- kernel 4: min-append + argmax
__global__ void finalize_kernel(const float* __restrict__ lws, float* __restrict__ out,
                                int rows, int n) {
    int i = blockIdx.x * blockDim.x + threadIdx.x;
    if (i >= rows) return;
    const float* lg = lws + (size_t)i * n;
    float mn = 3.4e38f, mxv = -3.4e38f; int am = 0;
    for (int k = 0; k < n; ++k) {
        float v = lg[k];
        out[(size_t)i * (n + 1) + k] = v;
        mn = fminf(mn, v);
        if (v > mxv) { mxv = v; am = k; }
    }
    float extra = mn - 1.0f;
    out[(size_t)i * (n + 1) + n] = extra;
    if (extra > mxv) am = n;
    out[(size_t)rows * (n + 1) + i] = (float)am;   // pred as float
}

// ---------------------------------------------------------------- host
extern "C" void kernel_launch(void* const* d_in, const int* in_sizes, int n_in,
                              void* d_out, int out_size, void* d_ws, size_t ws_size,
                              hipStream_t stream) {
    const float* support = (const float*)d_in[0];
    const float* query   = (const float*)d_in[1];
    const float* ln_g    = (const float*)d_in[2];
    const float* ln_b    = (const float*)d_in[3];
    const float* ln2_g   = (const float*)d_in[4];
    const float* ln2_b   = (const float*)d_in[5];
    const float* proj_W  = (const float*)d_in[6];
    const float* proj_b  = (const float*)d_in[7];
    const float* W1      = (const float*)d_in[8];
    const float* b1      = (const float*)d_in[9];
    const float* W2      = (const float*)d_in[10];
    const float* b2      = (const float*)d_in[11];
    const float* W3      = (const float*)d_in[12];
    const float* b3      = (const float*)d_in[13];

    const int N = 10, K = 1, Q = 5;            // per reference setup
    const int NK = N * K, NQ = N * Q;
    const int Bsz = in_sizes[0] / (NK * LL * DD);
    const int nS = Bsz * NK, nQr = Bsz * NQ;
    const int M  = Bsz * NQ * NK;

    char* ws = (char*)d_ws;
    float*    sn = (float*)ws;     ws += (size_t)nS  * LL * DD * 4;
    float*    qn = (float*)ws;     ws += (size_t)nQr * LL * DD * 4;
    _Float16* sh = (_Float16*)ws;  ws += (size_t)nS  * LL * DD * 2;
    _Float16* qh = (_Float16*)ws;  ws += (size_t)nQr * LL * DD * 2;
    _Float16* Wh = (_Float16*)ws;  ws += (size_t)DD * 4 * DD * 2;
    float*    lws = (float*)ws;

    ln_kernel<<<nS * LL, 256, 0, stream>>>(support, ln_g, ln_b, sn, sh);
    ln_kernel<<<nQr * LL, 256, 0, stream>>>(query, ln_g, ln_b, qn, qh);
    cvt_kernel<<<(DD * 4 * DD + 255) / 256, 256, 0, stream>>>(proj_W, Wh, DD * 4 * DD);

    size_t smem = (size_t)(LL * DD * 2) * 2          // s_h, q_h
                + (size_t)LL * LL * 4                // att
                + (size_t)LL * LL * 2 * 2            // p1, p2t
                + (size_t)LL * 1024 * 2              // enh
                + (size_t)LL * DD * 4                // pout
                + 512 * 4 * 2                        // pools
                + 256 * 4;                           // red
    pair_kernel<<<M, 256, smem, stream>>>(sn, qn, sh, qh, Wh, proj_b,
                                          ln2_g, ln2_b, W1, b1, W2, b2, W3, b3,
                                          lws, NQ, NK);

    finalize_kernel<<<(Bsz * NQ + 127) / 128, 128, 0, stream>>>(lws, (float*)d_out,
                                                                Bsz * NQ, N);
}
